// MSA_77695958385343
// MI455X (gfx1250) — matrix operand order repaired and checked by reading, not compiled, not run on hardware
//
#include <hip/hip_runtime.h>

#define DIMX   1024
#define INNERX 512
#define NBAT   4
#define NSEQ   2048
#define SCL    0.125f   // 1/sqrt(64)

#define BM 128
#define BN 64
#define BK 64

typedef __attribute__((ext_vector_type(16))) __bf16 v16bf;
typedef __attribute__((ext_vector_type(8)))  __bf16 v8bf;
typedef __attribute__((ext_vector_type(8)))  float  v8f;
typedef __attribute__((ext_vector_type(4)))  int    v4i;

typedef __attribute__((address_space(1))) v4i* gv4i_p;
typedef __attribute__((address_space(3))) v4i* lv4i_p;

__device__ __forceinline__ int lane_id() { return (int)(threadIdx.x & 31u); }

__device__ __forceinline__ unsigned enc_f(float f) {
  unsigned u = __float_as_uint(f);
  return (u & 0x80000000u) ? ~u : (u | 0x80000000u);
}
__device__ __forceinline__ float dec_f(unsigned e) {
  unsigned u = (e & 0x80000000u) ? (e & 0x7fffffffu) : ~e;
  return __uint_as_float(u);
}

__device__ __forceinline__ v8f wmma_bf16(v16bf a, v16bf b, v8f c) {
  return __builtin_amdgcn_wmma_f32_16x16x32_bf16(false, a, false, b, (short)0, c, false, false);
}

// ---- async global -> LDS (16B per lane), ASYNCcnt tracked ----
__device__ __forceinline__ void async_copy16(__bf16* lds, const __bf16* g) {
#if __has_builtin(__builtin_amdgcn_global_load_async_to_lds_b128)
  __builtin_amdgcn_global_load_async_to_lds_b128(
      (gv4i_p)(uintptr_t)g,
      (lv4i_p)(unsigned)(uintptr_t)lds,
      0, 0);
#else
  unsigned l32 = (unsigned)(uintptr_t)lds;
  unsigned long long g64 = (unsigned long long)(uintptr_t)g;
  asm volatile("global_load_async_to_lds_b128 %0, %1, off"
               :: "v"(l32), "v"(g64) : "memory");
#endif
}

__device__ __forceinline__ void wait_async0() {
#if __has_builtin(__builtin_amdgcn_s_wait_asynccnt)
  __builtin_amdgcn_s_wait_asynccnt(0);
#else
  asm volatile("s_wait_asynccnt 0x0" ::: "memory");
#endif
}

// A fragment 16x32 bf16 (lane m=lane&15, h=lane>>4):
// elems 0..7 : K = h*8 + 0..7 ; elems 8..15: K = 16 + h*8 + 0..7
__device__ __forceinline__ v16bf load_a_frag(const __bf16* A, int lda) {
  int lane = lane_id();
  int m = lane & 15, h = lane >> 4;
  const __bf16* base = A + (size_t)m * lda + h * 8;
  v8bf lo = *(const v8bf*)(base);
  v8bf hi = *(const v8bf*)(base + 16);
  v16bf out;
#pragma unroll
  for (int i = 0; i < 8; ++i) { out[i] = lo[i]; out[8 + i] = hi[i]; }
  return out;
}

// B fragment 32x16 bf16 from K-contiguous "Bt" (lane n=lane&15, h=lane>>4):
// elems 0..7 : K = h*8 + 0..7 ; elems 8..15: K = 16 + h*8 + 0..7
__device__ __forceinline__ v16bf load_b_frag(const __bf16* Bt, int ldb) {
  int lane = lane_id();
  int n = lane & 15, h = lane >> 4;
  const __bf16* base = Bt + (size_t)n * ldb + h * 8;
  v8bf lo = *(const v8bf*)(base);
  v8bf hi = *(const v8bf*)(base + 16);
  v16bf out;
#pragma unroll
  for (int i = 0; i < 8; ++i) { out[i] = lo[i]; out[8 + i] = hi[i]; }
  return out;
}

// stage loader: A tile BMxBK (1024 16B chunks), B tile BNxBK (512 chunks), 256 threads
__device__ __forceinline__ void load_stage(__bf16* sA, __bf16* sB,
                                           const __bf16* Ablk, const __bf16* Btblk,
                                           int k0, int lda, int ldb, int tid) {
#pragma unroll
  for (int t = 0; t < 4; ++t) {
    int c = tid + t * 256;
    int row = c >> 3, col = (c & 7) * 8;
    async_copy16(sA + row * BK + col, Ablk + (size_t)row * lda + k0 + col);
  }
#pragma unroll
  for (int t = 0; t < 2; ++t) {
    int c = tid + t * 256;
    int row = c >> 3, col = (c & 7) * 8;
    async_copy16(sB + row * BK + col, Btblk + (size_t)row * ldb + k0 + col);
  }
}

// Pipelined LDS-staged GEMM core: block = 256 thr (8 waves), block tile BMxBN,
// wave w computes rows [w*16, w*16+16) x BN. acc[4] = 4 n-frags of 16x16.
__device__ __forceinline__ void gemm_tile_lds(const __bf16* __restrict__ Ablk,
                                              const __bf16* __restrict__ Btblk,
                                              int K, int lda, int ldb, v8f acc[4]) {
  __shared__ __align__(16) __bf16 sA[2][BM * BK];
  __shared__ __align__(16) __bf16 sB[2][BN * BK];
  const int tid = threadIdx.x;
  const int wid = tid >> 5;
  const int m0w = wid * 16;

  load_stage(sA[0], sB[0], Ablk, Btblk, 0, lda, ldb, tid);
  wait_async0();
  __syncthreads();

  const int nst = K / BK;
  for (int s = 0; s < nst; ++s) {
    const int buf = s & 1;
    if (s + 1 < nst)
      load_stage(sA[buf ^ 1], sB[buf ^ 1], Ablk, Btblk, (s + 1) * BK, lda, ldb, tid);
#pragma unroll
    for (int kf = 0; kf < BK; kf += 32) {
      v16bf a = load_a_frag(&sA[buf][m0w * BK + kf], BK);
#pragma unroll
      for (int j = 0; j < 4; ++j) {
        v16bf b = load_b_frag(&sB[buf][(j * 16) * BK + kf], BK);
        acc[j] = wmma_bf16(a, b, acc[j]);
      }
    }
    wait_async0();
    __syncthreads();
  }
}

// ---------------- prep kernels ----------------

__global__ void k_init(unsigned* maxenc, float* sumz) {
  int t = threadIdx.x;
  if (t < NBAT) { maxenc[t] = 0u; sumz[t] = 0.f; }
}

__global__ void k_f32_to_bf16(const float* __restrict__ src, __bf16* __restrict__ dst, int n) {
  for (int i = blockIdx.x * blockDim.x + threadIdx.x; i < n; i += gridDim.x * blockDim.x)
    dst[i] = (__bf16)src[i];
}

// Wt[c*R + r] = W[r*C + c]
__global__ void k_transpose(const float* __restrict__ W, __bf16* __restrict__ Wt, int R, int C) {
  int total = R * C;
  for (int i = blockIdx.x * blockDim.x + threadIdx.x; i < total; i += gridDim.x * blockDim.x) {
    int c = i / R, r = i - c * R;
    Wt[i] = (__bf16)W[(size_t)r * C + c];
  }
}

// ---------------- GEMM kernels ----------------

// C = A * Bt^T ; mode 0: bf16 row-major; mode 1: per-batch transposed (vT planes)
__global__ __launch_bounds__(256)
void k_gemm_proj(const __bf16* __restrict__ A, const __bf16* __restrict__ Bt,
                 __bf16* __restrict__ C, int M, int N, int K, int mode) {
  const int nbc = N / BN;
  const int br = blockIdx.x / nbc, bc = blockIdx.x % nbc;
  v8f acc[4] = {};
  gemm_tile_lds(A + (size_t)br * BM * K, Bt + (size_t)bc * BN * K, K, K, K, acc);
  const int lane = lane_id(), n = lane & 15, h = lane >> 4;
  const int wid = threadIdx.x >> 5;
  const int m0 = br * BM + wid * 16, n0 = bc * BN;
  if (mode == 0) {
#pragma unroll
    for (int j = 0; j < 4; ++j) {
      int col = n0 + j * 16 + n;
#pragma unroll
      for (int r = 0; r < 8; ++r)
        C[(size_t)(m0 + 8 * h + r) * N + col] = (__bf16)acc[j][r];
    }
  } else {
    int bidx = m0 >> 11;
    int mloc0 = (m0 & (NSEQ - 1)) + 8 * h;
#pragma unroll
    for (int j = 0; j < 4; ++j) {
      int col = n0 + j * 16 + n;
      __bf16* dst = C + ((size_t)bidx * N + col) * (size_t)NSEQ + mloc0;
#pragma unroll
      for (int r = 0; r < 8; ++r) dst[r] = (__bf16)acc[j][r];
    }
  }
}

// S[b] = SCL * q[b]*k[b]^T (f32) + per-batch global max (monotone-uint atomicMax)
__global__ __launch_bounds__(256)
void k_gemm_s(const __bf16* __restrict__ q, const __bf16* __restrict__ kmat,
              float* __restrict__ S, unsigned* __restrict__ maxenc) {
  const int b = blockIdx.y;
  const int nbc = NSEQ / BN;  // 32
  const int br = blockIdx.x / nbc, bc = blockIdx.x % nbc;
  const __bf16* A  = q    + (size_t)b * NSEQ * INNERX + (size_t)br * BM * INNERX;
  const __bf16* Bt = kmat + (size_t)b * NSEQ * INNERX + (size_t)bc * BN * INNERX;
  float* Sb = S + (size_t)b * NSEQ * NSEQ;
  v8f acc[4] = {};
  gemm_tile_lds(A, Bt, INNERX, INNERX, INNERX, acc);
  const int lane = lane_id(), n = lane & 15, h = lane >> 4;
  const int wid = threadIdx.x >> 5;
  const int m0 = br * BM + wid * 16, n0 = bc * BN;
  float lmax = -3.4e38f;
#pragma unroll
  for (int j = 0; j < 4; ++j) {
    int col = n0 + j * 16 + n;
#pragma unroll
    for (int r = 0; r < 8; ++r) {
      float v = acc[j][r] * SCL;
      Sb[(size_t)(m0 + 8 * h + r) * NSEQ + col] = v;
      lmax = fmaxf(lmax, v);
    }
  }
  __shared__ float sred[256];
  int tid = threadIdx.x;
  sred[tid] = lmax;
  __syncthreads();
  for (int s = 128; s > 0; s >>= 1) {
    if (tid < s) sred[tid] = fmaxf(sred[tid], sred[tid + s]);
    __syncthreads();
  }
  if (tid == 0) atomicMax(&maxenc[b], enc_f(sred[0]));
}

// partial sums of exp(S - M) AND write P = bf16(exp(S - M))
__global__ __launch_bounds__(256)
void k_sumexp_writeP(const float* __restrict__ S, const unsigned* __restrict__ maxenc,
                     __bf16* __restrict__ P, float* __restrict__ partial) {
  const int b = blockIdx.y;
  const float Mb = dec_f(maxenc[b]);
  const float* Sb = S + (size_t)b * NSEQ * NSEQ;
  __bf16* Pb = P + (size_t)b * NSEQ * NSEQ;
  const int total = NSEQ * NSEQ;
  float local = 0.f;
  for (int i = blockIdx.x * blockDim.x + threadIdx.x; i < total; i += gridDim.x * blockDim.x) {
    float e = __expf(Sb[i] - Mb);
    Pb[i] = (__bf16)e;
    local += e;
  }
  __shared__ float sred[256];
  int tid = threadIdx.x;
  sred[tid] = local;
  __syncthreads();
  for (int s = 128; s > 0; s >>= 1) {
    if (tid < s) sred[tid] += sred[tid + s];
    __syncthreads();
  }
  if (tid == 0) partial[b * 1024 + blockIdx.x] = sred[0];
}

__global__ __launch_bounds__(256)
void k_reduce_sum(const float* __restrict__ partial, float* __restrict__ sumz) {
  const int b = blockIdx.x;
  __shared__ float sred[256];
  int tid = threadIdx.x;
  const float* p = partial + b * 1024;
  sred[tid] = p[tid] + p[tid + 256] + p[tid + 512] + p[tid + 768];
  __syncthreads();
  for (int s = 128; s > 0; s >>= 1) {
    if (tid < s) sred[tid] += sred[tid + s];
    __syncthreads();
  }
  if (tid == 0) sumz[b] = sred[0];
}

// SA[b] = (P[b] * v[b]) / Z  (bf16 out)
__global__ __launch_bounds__(256)
void k_gemm_sa(const __bf16* __restrict__ P, const __bf16* __restrict__ vT,
               __bf16* __restrict__ SA, const float* __restrict__ sumz) {
  const int b = blockIdx.y;
  const float invZ = 1.0f / sumz[b];
  const int nbc = INNERX / BN;  // 8
  const int br = blockIdx.x / nbc, bc = blockIdx.x % nbc;
  const __bf16* A  = P  + (size_t)b * NSEQ * NSEQ + (size_t)br * BM * NSEQ;
  const __bf16* Bt = vT + (size_t)b * INNERX * NSEQ + (size_t)bc * BN * NSEQ;
  __bf16* Cb = SA + (size_t)b * NSEQ * INNERX;
  v8f acc[4] = {};
  gemm_tile_lds(A, Bt, NSEQ, NSEQ, NSEQ, acc);
  const int lane = lane_id(), n = lane & 15, h = lane >> 4;
  const int wid = threadIdx.x >> 5;
  const int m0 = br * BM + wid * 16, n0 = bc * BN;
#pragma unroll
  for (int j = 0; j < 4; ++j) {
    int col = n0 + j * 16 + n;
#pragma unroll
    for (int r = 0; r < 8; ++r)
      Cb[(size_t)(m0 + 8 * h + r) * INNERX + col] = (__bf16)(acc[j][r] * invZ);
  }
}

// out = SA * Wo + bo (f32)
__global__ __launch_bounds__(256)
void k_gemm_out(const __bf16* __restrict__ SA, const __bf16* __restrict__ WoT,
                const float* __restrict__ bo, float* __restrict__ out) {
  const int nbc = DIMX / BN;  // 16
  const int br = blockIdx.x / nbc, bc = blockIdx.x % nbc;
  v8f acc[4] = {};
  gemm_tile_lds(SA + (size_t)br * BM * INNERX, WoT + (size_t)bc * BN * INNERX,
                INNERX, INNERX, INNERX, acc);
  const int lane = lane_id(), n = lane & 15, h = lane >> 4;
  const int wid = threadIdx.x >> 5;
  const int m0 = br * BM + wid * 16, n0 = bc * BN;
#pragma unroll
  for (int j = 0; j < 4; ++j) {
    int col = n0 + j * 16 + n;
    float bias = bo[col];
#pragma unroll
    for (int r = 0; r < 8; ++r)
      out[(size_t)(m0 + 8 * h + r) * DIMX + col] = acc[j][r] + bias;
  }
}

extern "C" void kernel_launch(void* const* d_in, const int* in_sizes, int n_in,
                              void* d_out, int out_size, void* d_ws, size_t ws_size,
                              hipStream_t stream) {
  (void)in_sizes; (void)n_in; (void)out_size; (void)ws_size;
  const float* input = (const float*)d_in[0];
  const float* Wq = (const float*)d_in[1];
  const float* Wk = (const float*)d_in[2];
  const float* Wv = (const float*)d_in[3];
  const float* Wo = (const float*)d_in[4];
  const float* bo = (const float*)d_in[5];
  float* out = (float*)d_out;

  char* ws = (char*)d_ws;
  size_t off = 0;
  auto take = [&](size_t bytes) -> char* {
    char* p = ws + off;
    off += (bytes + 255) & ~(size_t)255;
    return p;
  };
  __bf16* inputB = (__bf16*)take((size_t)NBAT * NSEQ * DIMX * 2);     // 16 MB
  __bf16* WqT    = (__bf16*)take((size_t)INNERX * DIMX * 2);          // 1 MB
  __bf16* WkT    = (__bf16*)take((size_t)INNERX * DIMX * 2);
  __bf16* WvT    = (__bf16*)take((size_t)INNERX * DIMX * 2);
  __bf16* WoT    = (__bf16*)take((size_t)DIMX * INNERX * 2);
  __bf16* qB     = (__bf16*)take((size_t)NBAT * NSEQ * INNERX * 2);   // 8 MB
  __bf16* kB     = (__bf16*)take((size_t)NBAT * NSEQ * INNERX * 2);
  __bf16* vT     = (__bf16*)take((size_t)NBAT * INNERX * NSEQ * 2);
  float*  Smat   = (float*) take((size_t)NBAT * NSEQ * NSEQ * 4);     // 64 MB
  __bf16* Pbf    = (__bf16*)take((size_t)NBAT * NSEQ * NSEQ * 2);     // 32 MB
  __bf16* SAb    = (__bf16*)take((size_t)NBAT * NSEQ * INNERX * 2);
  float*  partial= (float*) take((size_t)NBAT * 1024 * 4);
  unsigned* maxenc = (unsigned*)take(256);
  float*    sumz   = (float*)take(256);

  const int Mrows = NBAT * NSEQ;  // 8192

  k_init<<<1, 64, 0, stream>>>(maxenc, sumz);
  k_f32_to_bf16<<<4096, 256, 0, stream>>>(input, inputB, Mrows * DIMX);
  k_transpose<<<2048, 256, 0, stream>>>(Wq, WqT, DIMX, INNERX);
  k_transpose<<<2048, 256, 0, stream>>>(Wk, WkT, DIMX, INNERX);
  k_transpose<<<2048, 256, 0, stream>>>(Wv, WvT, DIMX, INNERX);
  k_transpose<<<2048, 256, 0, stream>>>(Wo, WoT, INNERX, DIMX);

  // blocks = (8192/128)*(512/64) = 64*8 = 512
  k_gemm_proj<<<512, 256, 0, stream>>>(inputB, WqT, qB, Mrows, INNERX, DIMX, 0);
  k_gemm_proj<<<512, 256, 0, stream>>>(inputB, WkT, kB, Mrows, INNERX, DIMX, 0);
  k_gemm_proj<<<512, 256, 0, stream>>>(inputB, WvT, vT, Mrows, INNERX, DIMX, 1);

  // per batch: (2048/128)*(2048/64) = 16*32 = 512
  k_gemm_s<<<dim3(512, 4), 256, 0, stream>>>(qB, kB, Smat, maxenc);

  k_sumexp_writeP<<<dim3(1024, 4), 256, 0, stream>>>(Smat, maxenc, Pbf, partial);
  k_reduce_sum<<<NBAT, 256, 0, stream>>>(partial, sumz);

  // per batch: (2048/128)*(512/64) = 16*8 = 128
  k_gemm_sa<<<dim3(128, 4), 256, 0, stream>>>(Pbf, vT, SAb, sumz);

  // (8192/128)*(1024/64) = 64*16 = 1024
  k_gemm_out<<<1024, 256, 0, stream>>>(SAb, WoT, bo, out);
}
